// MultiHeadAttention_32607391711801
// MI455X (gfx1250) — matrix operand order repaired and checked by reading, not compiled, not run on hardware
//
#include <hip/hip_runtime.h>
#include <hip/hip_bf16.h>
#include <math.h>

typedef __attribute__((ext_vector_type(16))) _Float16 v16h;
typedef __attribute__((ext_vector_type(8)))  _Float16 v8h;
typedef __attribute__((ext_vector_type(8)))  float    v8f;

#define EMBED    768
#define HEADS    12
#define HEAD_DIM 64
#define BATCH    16
#define SEQ      1024
#define BH       (BATCH * HEADS)        // 192
#define MROWS    (BATCH * SEQ)          // 16384
#define QKV_OUT  (3 * EMBED)            // 2304
#define SCALE_INV 0.036084391824351615f // 1/sqrt(768)

// ---------------------------------------------------------------------------
// WMMA helpers (contiguous k order: lane half h holds k = 16h..16h+15)
// ---------------------------------------------------------------------------
__device__ __forceinline__ v8f wmma16(v16h a, v16h b, v8f c) {
  return __builtin_amdgcn_wmma_f32_16x16x32_f16(false, a, false, b,
                                                (short)0, c, false, false);
}

// A fragment (16x32 f16, M x K): rowPtr = &A[m][kbase], m = lane&15.
// half 0: K = 0..15 ; half 1: K = 16..31.
__device__ __forceinline__ v16h load_a_frag(const _Float16* rowPtr, int half) {
  return *(const v16h*)(rowPtr + half * 16);
}

// B fragment (32x16 f16, K x N) read from a B-transposed [N][K] buffer:
// colPtr = &Bt[n][kbase], n = lane&15. b[j] = Bt[n][kbase + half*16 + j].
__device__ __forceinline__ v16h load_b_frag(const _Float16* colPtr, int half) {
  return *(const v16h*)(colPtr + half * 16);
}

// ---------------------------------------------------------------------------
// Kernel 0: f32 -> f16 conversion (grid-stride)
// ---------------------------------------------------------------------------
__global__ void __launch_bounds__(256) cvt_f32_to_f16(
    const float* __restrict__ s, _Float16* __restrict__ d, int n) {
  for (int i = blockIdx.x * blockDim.x + threadIdx.x; i < n;
       i += gridDim.x * blockDim.x)
    d[i] = (_Float16)s[i];
}

// ---------------------------------------------------------------------------
// Kernel 1: QKV projection  qkv = x @ w_qkv^T + b_qkv, fused re-layout.
// Each wave computes a 32x64 tile (MI=2 x NI=4) of the [16384 x 2304] result
// -> 8 f32 accumulators (64 VGPRs), no spills.
// Output column o decodes as o = h*192 + d*3 + s  (qkv index innermost).
// Q,K stored [bh][n][d] f16 ; V stored transposed [bh][d][n] f16.
// ---------------------------------------------------------------------------
__global__ void __launch_bounds__(256) qkv_gemm_kernel(
    const _Float16* __restrict__ xh,   // [16384][768]
    const _Float16* __restrict__ wqh,  // [2304][768]
    const float*    __restrict__ bqkv, // [2304]
    _Float16* __restrict__ Qb,
    _Float16* __restrict__ Kb,
    _Float16* __restrict__ Vtb) {
  const int TN   = QKV_OUT / 64;                       // 36
  const int wave = blockIdx.x * (blockDim.x >> 5) + (threadIdx.x >> 5);
  const int lane = threadIdx.x & 31;
  const int half = lane >> 4, l16 = lane & 15;
  const int tm = wave / TN, tn = wave % TN;
  const int m0 = tm * 32, n0 = tn * 64;

  v8f acc[2][4] = {};
  const _Float16* arow[2];
  const _Float16* brow[4];
#pragma unroll
  for (int i = 0; i < 2; ++i)
    arow[i] = xh + (size_t)(m0 + i * 16 + l16) * EMBED;
#pragma unroll
  for (int i = 0; i < 4; ++i)
    brow[i] = wqh + (size_t)(n0 + i * 16 + l16) * EMBED;

  for (int kb = 0; kb < EMBED; kb += 32) {
    v16h af[2], bf[4];
#pragma unroll
    for (int i = 0; i < 2; ++i) af[i] = load_a_frag(arow[i] + kb, half);
#pragma unroll
    for (int i = 0; i < 4; ++i) bf[i] = load_b_frag(brow[i] + kb, half);
#pragma unroll
    for (int mi = 0; mi < 2; ++mi)
#pragma unroll
      for (int ni = 0; ni < 4; ++ni)
        acc[mi][ni] = wmma16(af[mi], bf[ni], acc[mi][ni]);
  }

  const int b = m0 / SEQ;  // tiles never straddle batch (1024 % 32 == 0)
#pragma unroll
  for (int ni = 0; ni < 4; ++ni) {
    const int o   = n0 + ni * 16 + l16;
    const int h   = o / 192;
    const int rem = o % 192;
    const int d   = rem / 3;
    const int s   = rem % 3;
    const int bh  = b * HEADS + h;
    const float bias = bqkv[o];
#pragma unroll
    for (int mi = 0; mi < 2; ++mi) {
#pragma unroll
      for (int r = 0; r < 8; ++r) {
        const int m    = m0 + mi * 16 + r + 8 * half;
        const int nrow = m & (SEQ - 1);
        const _Float16 hv = (_Float16)(acc[mi][ni][r] + bias);
        if (s == 0)
          Qb[((size_t)bh * SEQ + nrow) * HEAD_DIM + d] = hv;
        else if (s == 1)
          Kb[((size_t)bh * SEQ + nrow) * HEAD_DIM + d] = hv;
        else
          Vtb[(size_t)bh * HEAD_DIM * SEQ + (size_t)d * SEQ + nrow] = hv;
      }
    }
  }
}

// ---------------------------------------------------------------------------
// Kernel 2: flash attention. One wave per (bh, 16-row q tile).
// Streams keys in tiles of 32; softmax with running (max, sum) per row.
// LDS stages S (f32) and P (f16 in A-fragment-friendly [m][k] layout).
// ---------------------------------------------------------------------------
__global__ void __launch_bounds__(256) attn_kernel(
    const _Float16* __restrict__ Qb,
    const _Float16* __restrict__ Kb,
    const _Float16* __restrict__ Vtb,
    _Float16* __restrict__ Ao) {  // [16384][768]
  __shared__ float    Sbuf[8][16 * 32];
  __shared__ _Float16 Pbuf[8][16 * 32];
  __shared__ float    rowf[8][16];

  const int wv   = threadIdx.x >> 5;
  const int lane = threadIdx.x & 31;
  const int half = lane >> 4, l16 = lane & 15;
  const int wave = blockIdx.x * 8 + wv;
  const int bh = wave >> 6;        // / (SEQ/16)
  const int qt = wave & 63;
  const int q0 = qt * 16;

  const _Float16* Qh = Qb  + (size_t)bh * SEQ * HEAD_DIM;
  const _Float16* Kh = Kb  + (size_t)bh * SEQ * HEAD_DIM;
  const _Float16* Vh = Vtb + (size_t)bh * HEAD_DIM * SEQ;

  const _Float16* qrow = Qh + (size_t)(q0 + l16) * HEAD_DIM;
  const v16h qa0 = load_a_frag(qrow, half);        // K = 0..31 of D
  const v16h qa1 = load_a_frag(qrow + 32, half);   // K = 32..63 of D

  v8f o[4] = {};
  float mrow = -3.0e38f, lrow = 0.0f;              // valid in lanes 0..15

  for (int kt = 0; kt < SEQ; kt += 32) {
    // ---- S = (Q K^T) * scale for 32 keys (two 16-col tiles) ----
    v8f s0 = {}, s1 = {};
    {
      const _Float16* kr0 = Kh + (size_t)(kt + l16) * HEAD_DIM;
      const _Float16* kr1 = Kh + (size_t)(kt + 16 + l16) * HEAD_DIM;
      s0 = wmma16(qa0, load_b_frag(kr0, half), s0);
      s0 = wmma16(qa1, load_b_frag(kr0 + 32, half), s0);
      s1 = wmma16(qa0, load_b_frag(kr1, half), s1);
      s1 = wmma16(qa1, load_b_frag(kr1 + 32, half), s1);
    }
#pragma unroll
    for (int r = 0; r < 8; ++r) {
      const int row = r + 8 * half;
      Sbuf[wv][row * 32 + l16]      = s0[r] * SCALE_INV;
      Sbuf[wv][row * 32 + 16 + l16] = s1[r] * SCALE_INV;
    }
    __syncthreads();

    // ---- per-row streaming softmax (lane r owns row r) ----
    if (lane < 16) {
      const float* srow = &Sbuf[wv][lane * 32];
      float mx = mrow;
#pragma unroll
      for (int j = 0; j < 32; ++j) mx = fmaxf(mx, srow[j]);
      const float alpha = __expf(mrow - mx);
      float sum = 0.0f;
#pragma unroll
      for (int j = 0; j < 32; ++j) {
        const float p = __expf(srow[j] - mx);
        sum += p;
        Pbuf[wv][lane * 32 + j] = (_Float16)p;
      }
      lrow = lrow * alpha + sum;
      mrow = mx;
      rowf[wv][lane] = alpha;
    }
    __syncthreads();

    // ---- rescale O and accumulate P @ V ----
    float al[8];
#pragma unroll
    for (int r = 0; r < 8; ++r) al[r] = rowf[wv][8 * half + r];
    const v16h pa = load_a_frag(&Pbuf[wv][l16 * 32], half);
#pragma unroll
    for (int d = 0; d < 4; ++d) {
      const v16h vb =
          load_b_frag(Vh + (size_t)(d * 16 + l16) * SEQ + kt, half);
#pragma unroll
      for (int r = 0; r < 8; ++r) o[d][r] *= al[r];
      o[d] = wmma16(pa, vb, o[d]);
    }
  }

  // ---- finalize: divide by row sums, write [b][n][h*64+d] f16 ----
  if (lane < 16) rowf[wv][lane] = 1.0f / lrow;
  __syncthreads();
  float linv[8];
#pragma unroll
  for (int r = 0; r < 8; ++r) linv[r] = rowf[wv][8 * half + r];

  const int b = bh / HEADS, h = bh % HEADS;
#pragma unroll
  for (int d = 0; d < 4; ++d) {
#pragma unroll
    for (int r = 0; r < 8; ++r) {
      const int row = q0 + r + 8 * half;
      const int col = h * HEAD_DIM + d * 16 + l16;
      Ao[((size_t)b * SEQ + row) * EMBED + col] = (_Float16)(o[d][r] * linv[r]);
    }
  }
}

// ---------------------------------------------------------------------------
// Kernel 3: output projection  out = A @ w_proj^T + b_proj  (f32 output)
// Per-wave 32x64 tile (MI=2 x NI=4), no spills.
// ---------------------------------------------------------------------------
__global__ void __launch_bounds__(256) proj_gemm_kernel(
    const _Float16* __restrict__ Ah,  // [16384][768]
    const _Float16* __restrict__ wph, // [768][768]
    const float*    __restrict__ bp,  // [768]
    float* __restrict__ out) {
  const int TN   = EMBED / 64;                         // 12
  const int wave = blockIdx.x * (blockDim.x >> 5) + (threadIdx.x >> 5);
  const int lane = threadIdx.x & 31;
  const int half = lane >> 4, l16 = lane & 15;
  const int tm = wave / TN, tn = wave % TN;
  const int m0 = tm * 32, n0 = tn * 64;

  v8f acc[2][4] = {};
  const _Float16* arow[2];
  const _Float16* brow[4];
#pragma unroll
  for (int i = 0; i < 2; ++i)
    arow[i] = Ah + (size_t)(m0 + i * 16 + l16) * EMBED;
#pragma unroll
  for (int i = 0; i < 4; ++i)
    brow[i] = wph + (size_t)(n0 + i * 16 + l16) * EMBED;

  for (int kb = 0; kb < EMBED; kb += 32) {
    v16h af[2], bf[4];
#pragma unroll
    for (int i = 0; i < 2; ++i) af[i] = load_a_frag(arow[i] + kb, half);
#pragma unroll
    for (int i = 0; i < 4; ++i) bf[i] = load_b_frag(brow[i] + kb, half);
#pragma unroll
    for (int mi = 0; mi < 2; ++mi)
#pragma unroll
      for (int ni = 0; ni < 4; ++ni)
        acc[mi][ni] = wmma16(af[mi], bf[ni], acc[mi][ni]);
  }

#pragma unroll
  for (int ni = 0; ni < 4; ++ni) {
    const int n = n0 + ni * 16 + l16;
    const float bias = bp[n];
#pragma unroll
    for (int mi = 0; mi < 2; ++mi) {
#pragma unroll
      for (int r = 0; r < 8; ++r) {
        const int m = m0 + mi * 16 + r + 8 * half;
        out[(size_t)m * EMBED + n] = acc[mi][ni][r] + bias;
      }
    }
  }
}

// ---------------------------------------------------------------------------
// Host launcher
// ---------------------------------------------------------------------------
extern "C" void kernel_launch(void* const* d_in, const int* in_sizes, int n_in,
                              void* d_out, int out_size, void* d_ws,
                              size_t ws_size, hipStream_t stream) {
  const float* x      = (const float*)d_in[0];
  const float* w_qkv  = (const float*)d_in[1];
  const float* b_qkv  = (const float*)d_in[2];
  const float* w_proj = (const float*)d_in[3];
  const float* b_proj = (const float*)d_in[4];
  float* out = (float*)d_out;

  // workspace carve-up (f16 buffers), 256B aligned sizes
  char* ws = (char*)d_ws;
  size_t off = 0;
  auto alloc = [&](size_t bytes) {
    void* p = ws + off;
    off += (bytes + 255) & ~(size_t)255;
    return p;
  };
  const size_t nX  = (size_t)MROWS * EMBED;        // 12.58M
  const size_t nWQ = (size_t)QKV_OUT * EMBED;      // 1.77M
  const size_t nWP = (size_t)EMBED * EMBED;        // 0.59M
  const size_t nHD = (size_t)BH * SEQ * HEAD_DIM;  // 12.58M

  _Float16* xh  = (_Float16*)alloc(nX * 2);
  _Float16* wqh = (_Float16*)alloc(nWQ * 2);
  _Float16* wph = (_Float16*)alloc(nWP * 2);
  _Float16* Qb  = (_Float16*)alloc(nHD * 2);
  _Float16* Kb  = (_Float16*)alloc(nHD * 2);
  _Float16* Vtb = (_Float16*)alloc(nHD * 2);
  _Float16* Ao  = (_Float16*)alloc(nX * 2);
  (void)ws_size;

  // 0) converts
  cvt_f32_to_f16<<<2048, 256, 0, stream>>>(x, xh, (int)nX);
  cvt_f32_to_f16<<<(int)((nWQ + 255) / 256), 256, 0, stream>>>(w_qkv, wqh, (int)nWQ);
  cvt_f32_to_f16<<<(int)((nWP + 255) / 256), 256, 0, stream>>>(w_proj, wph, (int)nWP);

  // 1) QKV GEMM: (16384/32) * (2304/64) = 18432 waves -> 2304 blocks of 8 waves
  qkv_gemm_kernel<<<2304, 256, 0, stream>>>(xh, wqh, b_qkv, Qb, Kb, Vtb);

  // 2) attention: 192 bh * 64 q-tiles = 12288 waves -> 1536 blocks
  attn_kernel<<<1536, 256, 0, stream>>>(Qb, Kb, Vtb, Ao);

  // 3) projection: (16384/32) * (768/64) = 6144 waves -> 768 blocks
  proj_gemm_kernel<<<768, 256, 0, stream>>>(Ao, wph, b_proj, out);
}
